// LRNet_CIFAR10_ver2X_36009005810028
// MI455X (gfx1250) — compile-verified, matrix operand order checked by reading, not run on hardware
//
#include <hip/hip_runtime.h>
#include <hip/hip_bf16.h>
#include <math.h>

typedef _Float16 h16;
typedef __attribute__((ext_vector_type(16))) _Float16 v16h;
typedef __attribute__((ext_vector_type(8)))  float    v8f;

#define EPSC 1e-10f
#define BN_EPSC 1e-5f

// ---------------------------------------------------------------------------
// Tensor Data Mover: 2-D tile load global->LDS via D# descriptor.
// Group0: [1:0]=count=1, [63:32]=lds_addr, [120:64]=global_addr, [127:126]=2.
// Group1: [17:16]=data_size(1->2B), [79:48]=tensor_dim0, [111:80]=tensor_dim1,
//         [127:112]=tile_dim0, [143:128]=tile_dim1, [207:160]=dim0_stride.
// ---------------------------------------------------------------------------
#if defined(__has_builtin)
#if __has_builtin(__builtin_amdgcn_tensor_load_to_lds)
#define HAVE_TDM 1
#endif
#endif

#ifdef HAVE_TDM
typedef __attribute__((ext_vector_type(4))) unsigned int u32x4;
typedef __attribute__((ext_vector_type(8))) int i32x8;
typedef __attribute__((ext_vector_type(4))) int i32x4;

__device__ __forceinline__ void tdm_load_2d(unsigned lds_off, const void* gptr,
                                            unsigned tile_d0, unsigned tile_d1,
                                            unsigned tensor_d0, unsigned tensor_d1,
                                            unsigned stride_d0 /*elements*/) {
  unsigned long long ga = (unsigned long long)(size_t)gptr;
  u32x4 g0;
  g0[0] = 1u;                                           // count=1, user mode
  g0[1] = lds_off;                                      // lds_addr (bytes)
  g0[2] = (unsigned)(ga & 0xffffffffu);                 // global_addr[31:0]
  g0[3] = (unsigned)((ga >> 32) & 0x01ffffffu) | (2u << 30);  // [56:32]|type=2
  i32x8 g1;
  g1[0] = (int)(1u << 16);                              // data_size=1 (2 bytes)
  g1[1] = (int)((tensor_d0 & 0xffffu) << 16);           // tensor_dim0[15:0]
  g1[2] = (int)((tensor_d0 >> 16) | ((tensor_d1 & 0xffffu) << 16));
  g1[3] = (int)((tensor_d1 >> 16) | (tile_d0 << 16));   // tile_dim0
  g1[4] = (int)(tile_d1 & 0xffffu);                     // tile_dim1, tile_dim2=0
  g1[5] = (int)stride_d0;                               // dim0_stride[31:0]
  g1[6] = 0;                                            // stride[47:32], dim1_stride=0
  g1[7] = 0;
  i32x4 z4v = {0, 0, 0, 0};
#if __clang_major__ >= 23
  i32x8 z8v = {0, 0, 0, 0, 0, 0, 0, 0};
  __builtin_amdgcn_tensor_load_to_lds(g0, g1, z4v, z4v, z8v, 0);
#else
  __builtin_amdgcn_tensor_load_to_lds(g0, g1, z4v, z4v, 0);
#endif
}
#endif  // HAVE_TDM

// ---------------------------------------------------------------------------
// Fragment loader for 16-bit WMMA A/B operands (ISA 7.12.2 layout):
// lanes 0-15 hold K {0..7,16..23}, lanes 16-31 hold K {8..15,24..31} of their
// row/col (lane%16). LDS row stride 32 halves = 64B keeps runs 16B aligned.
// ---------------------------------------------------------------------------
__device__ __forceinline__ v16h ld_frag(const h16* __restrict__ base, int lane) {
  const int koff = (lane & 16) ? 8 : 0;
  v16h f;
#pragma unroll
  for (int q = 0; q < 8; ++q) f[q] = base[koff + q];
#pragma unroll
  for (int q = 0; q < 8; ++q) f[8 + q] = base[16 + koff + q];
  return f;
}

// sign_gumbel: p = P(N(m,v)>0); binary-concrete, TAU=1 -> tanh(t/2)
__device__ __forceinline__ float sgum(float m, float v, float u) {
  float p = 0.5f * (1.0f + erff(m * rsqrtf(2.0f * v + EPSC)));
  p = fminf(fmaxf(p, EPSC), 1.0f - EPSC);
  float logits = logf(p) - log1pf(-p);
  float uc = fminf(fmaxf(u, EPSC), 1.0f - EPSC);
  float L = logf(uc) - log1pf(-uc);
  return tanhf((logits + L) * 0.5f);
}

// ---------------------------------------------------------------------------
// Implicit-GEMM lrconv (compile-time dims): D_m = Wm*patch(z), D_v = Wv*patch^2
// M = COUT (tile 64), N = B*H*W pixels (tile 32), K = CIN*9 padded to KPAD.
// MODE 0: fused sign_gumbel -> zout f16. MODE 1: write m,v f32.
// ---------------------------------------------------------------------------
template <int CIN, int COUT, int HH, int WW, int KPAD, int MODE>
__global__ __launch_bounds__(256)
void lrconv_wmma(const h16* __restrict__ zin, const h16* __restrict__ Wm,
                 const h16* __restrict__ Wv, const float* __restrict__ bias,
                 const float* __restrict__ gain, const float* __restrict__ u,
                 float* __restrict__ mout, float* __restrict__ vout,
                 h16* __restrict__ zout) {
  constexpr int HW = HH * WW;
  __shared__ h16 sAm[64 * 32];
  __shared__ h16 sAv[64 * 32];
  __shared__ h16 sB[32 * 32];     // transposed: [n][k]

  const int tid  = threadIdx.x;
  const int lane = tid & 31;
  const int wave = tid >> 5;
  const int msub = wave >> 1;     // 0..3
  const int nsub = wave & 1;      // 0..1
  const int ntile = blockIdx.x;
  const int mtile = blockIdx.y;

  v8f accm = {};
  v8f accv = {};

  // B-tile (im2col) builder coords: thread -> (n-local, 4 K values)
  const int nlB  = tid >> 3;       // 0..31
  const int k0B  = (tid & 7) * 4;  // 0,4,..,28
  const int npix = ntile * 32 + nlB;
  const int bB   = npix / HW;      // power-of-two -> shift
  const int remB = npix & (HW - 1);
  const int yB   = remB / WW;
  const int xB   = remB & (WW - 1);

#ifndef HAVE_TDM
  const int rowA = tid >> 2;       // 0..63
  const int cgA  = (tid & 3) * 8;  // 0,8,16,24
  const size_t gA = (size_t)(mtile * 64 + rowA) * KPAD + cgA;
#endif

  constexpr int KSTEPS = KPAD / 32;
  for (int kt = 0; kt < KSTEPS; ++kt) {
    __syncthreads();               // previous fragments consumed
#ifdef HAVE_TDM
    if (wave == 0) {               // TDM DMA: 64x32-half weight tiles -> LDS
      const size_t goff = ((size_t)mtile * 64 * KPAD + (size_t)kt * 32) * 2;
      tdm_load_2d((unsigned)(size_t)(void*)sAm, (const char*)Wm + goff,
                  32, 64, KPAD, COUT, KPAD);
      tdm_load_2d((unsigned)(size_t)(void*)sAv, (const char*)Wv + goff,
                  32, 64, KPAD, COUT, KPAD);
    }
#else
#pragma unroll
    for (int i = 0; i < 8; ++i) {
      sAm[rowA * 32 + cgA + i] = Wm[gA + (size_t)kt * 32 + i];
      sAv[rowA * 32 + cgA + i] = Wv[gA + (size_t)kt * 32 + i];
    }
#endif
    // patch tile (3x3 SAME halo, zero pad; also zero-pads K for conv1)
#pragma unroll
    for (int i = 0; i < 4; ++i) {
      int k  = kt * 32 + k0B + i;
      int ci = k / 9;
      h16 val = (h16)0.0f;
      if (ci < CIN) {
        int rs = k - ci * 9;
        int yy = yB + rs / 3 - 1;
        int xx = xB + rs % 3 - 1;
        if (yy >= 0 && yy < HH && xx >= 0 && xx < WW)
          val = zin[((size_t)(bB * CIN + ci) * HH + yy) * WW + xx];
      }
      sB[nlB * 32 + k0B + i] = val;
    }
#ifdef HAVE_TDM
    if (wave == 0) __builtin_amdgcn_s_wait_tensorcnt(0);
#endif
    __syncthreads();

    v16h fam = ld_frag(&sAm[(msub * 16 + (lane & 15)) * 32], lane);
    v16h fav = ld_frag(&sAv[(msub * 16 + (lane & 15)) * 32], lane);
    v16h fb  = ld_frag(&sB[(nsub * 16 + (lane & 15)) * 32], lane);
    v16h fb2;                      // squared patch fragment (v_pk_mul_f16)
#pragma unroll
    for (int i = 0; i < 16; ++i) fb2[i] = fb[i] * fb[i];

    accm = __builtin_amdgcn_wmma_f32_16x16x32_f16(false, fam, false, fb,
                                                  (short)0, accm, false, false);
    accv = __builtin_amdgcn_wmma_f32_16x16x32_f16(false, fav, false, fb2,
                                                  (short)0, accv, false, false);
  }

  // Epilogue: C/D layout -> M = r + 8*(lane>=16), N = lane%16
  const int crow = mtile * 64 + msub * 16 + ((lane & 16) ? 8 : 0);
  const int ncol = ntile * 32 + nsub * 16 + (lane & 15);
  const int b = ncol / HW;
  const int rem = ncol & (HW - 1);
  const int y = rem / WW;
  const int x = rem & (WW - 1);
#pragma unroll
  for (int r = 0; r < 8; ++r) {
    int c = crow + r;
    float g = gain[c];
    float mval = accm[r] * g + bias[c] * g;     // (conv+bias)*gain
    float vval = fmaxf(accv[r] * g * g, EPSC);
    size_t idx = ((size_t)(b * COUT + c) * HH + y) * WW + x;
    if (MODE == 0) {
      zout[idx] = (h16)sgum(mval, vval, u[idx]);
    } else {
      mout[idx] = mval;
      vout[idx] = vval;
    }
  }
}

// avgpool(m), avgpool(v)/4, sign_gumbel -> f16
__global__ void pool_sign_kernel(const float* __restrict__ m,
                                 const float* __restrict__ v,
                                 const float* __restrict__ u,
                                 h16* __restrict__ zout, int C, int H, int W) {
  int Ho = H >> 1, Wo = W >> 1;
  size_t total = (size_t)128 * C * Ho * Wo;
  size_t idx = (size_t)blockIdx.x * 256 + threadIdx.x;
  if (idx >= total) return;
  int xo = (int)(idx % Wo);
  size_t t = idx / Wo;
  int yo = (int)(t % Ho); t /= Ho;
  int c  = (int)(t % C);
  int b  = (int)(t / C);
  size_t base = ((size_t)(b * C + c) * H + yo * 2) * W + xo * 2;
  float ms = m[base] + m[base + 1] + m[base + W] + m[base + W + 1];
  float vs = v[base] + v[base + 1] + v[base + W] + v[base + W + 1];
  zout[idx] = (h16)sgum(ms * 0.25f, vs * (1.0f / 16.0f), u[idx]);
}

// conv6: out = m + sqrt(v)*eps, 2x2 avgpool -> [128,512,4,4] f32
__global__ void sample_pool_kernel(const float* __restrict__ m,
                                   const float* __restrict__ v,
                                   const float* __restrict__ eps,
                                   float* __restrict__ out6) {
  size_t total = (size_t)128 * 512 * 16;
  size_t idx = (size_t)blockIdx.x * 256 + threadIdx.x;
  if (idx >= total) return;
  int xo = (int)(idx & 3);
  size_t t = idx >> 2;
  int yo = (int)(t & 3); t >>= 2;
  int c  = (int)(t & 511);
  int b  = (int)(t >> 9);
  size_t base = ((size_t)(b * 512 + c) * 8 + yo * 2) * 8 + xo * 2;
  float s = 0.0f;
#pragma unroll
  for (int dy = 0; dy < 2; ++dy)
#pragma unroll
    for (int dx = 0; dx < 2; ++dx) {
      size_t i = base + dy * 8 + dx;
      s += m[i] + sqrtf(v[i]) * eps[i];
    }
  out6[idx] = s * 0.25f;
}

// BN over (B,H,W) per channel + ReLU -> f16 [B, 8192]
__global__ __launch_bounds__(256)
void bn_relu_kernel(const float* __restrict__ out6,
                    const float* __restrict__ gamma,
                    const float* __restrict__ beta, h16* __restrict__ a6) {
  __shared__ float ssum[256];
  __shared__ float ssq[256];
  int c = blockIdx.x;
  int tid = threadIdx.x;
  float s = 0.0f, q = 0.0f;
  for (int i = tid; i < 2048; i += 256) {
    int b = i >> 4, p = i & 15;
    float v = out6[((size_t)b * 512 + c) * 16 + p];
    s += v; q += v * v;
  }
  ssum[tid] = s; ssq[tid] = q;
  __syncthreads();
  for (int off = 128; off > 0; off >>= 1) {
    if (tid < off) { ssum[tid] += ssum[tid + off]; ssq[tid] += ssq[tid + off]; }
    __syncthreads();
  }
  float mean = ssum[0] * (1.0f / 2048.0f);
  float var  = ssq[0] * (1.0f / 2048.0f) - mean * mean;
  float scale = gamma[c] * rsqrtf(var + BN_EPSC);
  float shift = beta[c] - mean * scale;
  for (int i = tid; i < 2048; i += 256) {
    int b = i >> 4, p = i & 15;
    float v = out6[((size_t)b * 512 + c) * 16 + p];
    v = fmaxf(v * scale + shift, 0.0f);
    a6[(size_t)b * 8192 + c * 16 + p] = (h16)v;
  }
}

// fc1 WMMA GEMM: [128,8192]x[8192,1024] + bias, ReLU -> f16
__global__ __launch_bounds__(256)
void fc1_wmma(const h16* __restrict__ A, const h16* __restrict__ Bm,
              const float* __restrict__ bias, h16* __restrict__ out) {
  __shared__ h16 sA[32 * 32];
  __shared__ h16 sB[64 * 32];     // transposed [n][k]
  const int tid = threadIdx.x, lane = tid & 31, wave = tid >> 5;
  const int msub = wave >> 2;     // 0..1
  const int nsub = wave & 3;      // 0..3
  const int m0 = blockIdx.y * 32;
  const int n0 = blockIdx.x * 64;
  v8f acc = {};
  const int rowA = tid >> 3, kgA = (tid & 7) * 4;
  const int nlB  = tid >> 2, kgB = (tid & 3) * 8;
  for (int k0 = 0; k0 < 8192; k0 += 32) {
    __syncthreads();
#pragma unroll
    for (int i = 0; i < 4; ++i)
      sA[rowA * 32 + kgA + i] = A[(size_t)(m0 + rowA) * 8192 + k0 + kgA + i];
#pragma unroll
    for (int i = 0; i < 8; ++i)
      sB[nlB * 32 + kgB + i] = Bm[(size_t)(k0 + kgB + i) * 1024 + n0 + nlB];
    __syncthreads();
    v16h fa = ld_frag(&sA[(msub * 16 + (lane & 15)) * 32], lane);
    v16h fb = ld_frag(&sB[(nsub * 16 + (lane & 15)) * 32], lane);
    acc = __builtin_amdgcn_wmma_f32_16x16x32_f16(false, fa, false, fb,
                                                 (short)0, acc, false, false);
  }
  const int row = m0 + msub * 16 + ((lane & 16) ? 8 : 0);
  const int col = n0 + nsub * 16 + (lane & 15);
#pragma unroll
  for (int r = 0; r < 8; ++r) {
    float v = fmaxf(acc[r] + bias[col], 0.0f);
    out[(size_t)(row + r) * 1024 + col] = (h16)v;
  }
}

// fc2: [128,1024]x[1024,10] + bias (tiny; VALU)
__global__ void fc2_kernel(const h16* __restrict__ a7,
                           const float* __restrict__ W2,
                           const float* __restrict__ b2,
                           float* __restrict__ out) {
  int idx = blockIdx.x * 256 + threadIdx.x;
  if (idx >= 128 * 10) return;
  int b = idx / 10, j = idx - b * 10;
  float s = b2[j];
  for (int k = 0; k < 1024; ++k)
    s += (float)a7[(size_t)b * 1024 + k] * W2[(size_t)k * 10 + j];
  out[idx] = s;
}

// Ternary weight statistics: mu = 2*p1+p0-1, var = (1-p0) - mu^2 -> f16
__global__ void prep_w_kernel(const float* __restrict__ alpha,
                              const float* __restrict__ betta,
                              h16* __restrict__ Wm, h16* __restrict__ Wv,
                              int Cout, int Cin, int Kpad) {
  int idx = blockIdx.x * 256 + threadIdx.x;
  if (idx >= Cout * Kpad) return;
  int c = idx / Kpad, k = idx - c * Kpad;
  float mu = 0.0f, var = 0.0f;
  if (k < Cin * 9) {
    float a  = alpha[(size_t)c * Cin * 9 + k];
    float bt = betta[(size_t)c * Cin * 9 + k];
    float p0 = 1.0f / (1.0f + expf(-a));
    float p1 = (1.0f / (1.0f + expf(-bt))) * (1.0f - p0);
    mu = 2.0f * p1 + p0 - 1.0f;
    var = (1.0f - p0) - mu * mu;
  }
  Wm[idx] = (h16)mu;
  Wv[idx] = (h16)var;
}

__global__ void f32_to_f16_kernel(const float* __restrict__ in,
                                  h16* __restrict__ out, size_t n) {
  size_t i = (size_t)blockIdx.x * 256 + threadIdx.x;
  if (i < n) out[i] = (h16)in[i];
}

// ---------------------------------------------------------------------------
extern "C" void kernel_launch(void* const* d_in, const int* in_sizes, int n_in,
                              void* d_out, int out_size, void* d_ws, size_t ws_size,
                              hipStream_t stream) {
  (void)in_sizes; (void)n_in; (void)out_size; (void)ws_size;
  const float* x    = (const float*)d_in[0];
  const float* u1   = (const float*)d_in[1];
  const float* u2   = (const float*)d_in[2];
  const float* u3   = (const float*)d_in[3];
  const float* u4   = (const float*)d_in[4];
  const float* u5   = (const float*)d_in[5];
  const float* eps6 = (const float*)d_in[6];
  const float *al[6], *be[6], *bi[6], *ga[6];
  int pi = 7;
  for (int l = 0; l < 6; ++l) {
    al[l] = (const float*)d_in[pi++];
    be[l] = (const float*)d_in[pi++];
    bi[l] = (const float*)d_in[pi++];
    ga[l] = (const float*)d_in[pi++];
  }
  const float* gamma = (const float*)d_in[31];
  const float* beta  = (const float*)d_in[32];
  const float* W1    = (const float*)d_in[33];
  const float* b1    = (const float*)d_in[34];
  const float* W2    = (const float*)d_in[35];
  const float* b2    = (const float*)d_in[36];

  char* ws = (char*)d_ws;
  size_t off = 0;
  auto alloc = [&](size_t bytes) -> void* {
    void* p = ws + off;
    off = (off + bytes + 255) & ~(size_t)255;
    return p;
  };

  const int Cin[6] = {3, 128, 128, 256, 256, 512};
  const int Co[6]  = {128, 128, 256, 256, 512, 512};
  const int Kp[6]  = {32, 1152, 1152, 2304, 2304, 4608};

  h16* xh = (h16*)alloc((size_t)393216 * 2);
  h16* z1 = (h16*)alloc((size_t)16777216 * 2);
  h16* z2 = (h16*)alloc((size_t)4194304 * 2);
  h16* z3 = (h16*)alloc((size_t)8388608 * 2);
  h16* z4 = (h16*)alloc((size_t)2097152 * 2);
  h16* z5 = (h16*)alloc((size_t)4194304 * 2);
  float* mbuf = (float*)alloc((size_t)16777216 * 4);  // reused per MV layer
  float* vbuf = (float*)alloc((size_t)16777216 * 4);
  h16 *Wm[6], *Wv[6];
  for (int l = 0; l < 6; ++l) {
    Wm[l] = (h16*)alloc((size_t)Co[l] * Kp[l] * 2);
    Wv[l] = (h16*)alloc((size_t)Co[l] * Kp[l] * 2);
  }
  float* out6 = (float*)alloc((size_t)128 * 512 * 16 * 4);
  h16* a6  = (h16*)alloc((size_t)128 * 8192 * 2);
  h16* W1h = (h16*)alloc((size_t)8388608 * 2);
  h16* a7  = (h16*)alloc((size_t)131072 * 2);

  // --- prep ---
  f32_to_f16_kernel<<<(393216 + 255) / 256, 256, 0, stream>>>(x, xh, 393216);
  f32_to_f16_kernel<<<(8388608 + 255) / 256, 256, 0, stream>>>(W1, W1h, 8388608);
  for (int l = 0; l < 6; ++l) {
    int n = Co[l] * Kp[l];
    prep_w_kernel<<<(n + 255) / 256, 256, 0, stream>>>(al[l], be[l], Wm[l], Wv[l],
                                                       Co[l], Cin[l], Kp[l]);
  }

  // --- conv stack (per-layer template instantiations) ---
  lrconv_wmma<3, 128, 32, 32, 32, 0><<<dim3(4096, 2), 256, 0, stream>>>(
      xh, Wm[0], Wv[0], bi[0], ga[0], u1, nullptr, nullptr, z1);

  lrconv_wmma<128, 128, 32, 32, 1152, 1><<<dim3(4096, 2), 256, 0, stream>>>(
      z1, Wm[1], Wv[1], bi[1], ga[1], nullptr, mbuf, vbuf, nullptr);
  {
    size_t n = (size_t)128 * 128 * 16 * 16;
    pool_sign_kernel<<<(unsigned)((n + 255) / 256), 256, 0, stream>>>(
        mbuf, vbuf, u2, z2, 128, 32, 32);
  }

  lrconv_wmma<128, 256, 16, 16, 1152, 0><<<dim3(1024, 4), 256, 0, stream>>>(
      z2, Wm[2], Wv[2], bi[2], ga[2], u3, nullptr, nullptr, z3);

  lrconv_wmma<256, 256, 16, 16, 2304, 1><<<dim3(1024, 4), 256, 0, stream>>>(
      z3, Wm[3], Wv[3], bi[3], ga[3], nullptr, mbuf, vbuf, nullptr);
  {
    size_t n = (size_t)128 * 256 * 8 * 8;
    pool_sign_kernel<<<(unsigned)((n + 255) / 256), 256, 0, stream>>>(
        mbuf, vbuf, u4, z4, 256, 16, 16);
  }

  lrconv_wmma<256, 512, 8, 8, 2304, 0><<<dim3(256, 8), 256, 0, stream>>>(
      z4, Wm[4], Wv[4], bi[4], ga[4], u5, nullptr, nullptr, z5);

  lrconv_wmma<512, 512, 8, 8, 4608, 1><<<dim3(256, 8), 256, 0, stream>>>(
      z5, Wm[5], Wv[5], bi[5], ga[5], nullptr, mbuf, vbuf, nullptr);
  {
    size_t n = (size_t)128 * 512 * 16;
    sample_pool_kernel<<<(unsigned)((n + 255) / 256), 256, 0, stream>>>(
        mbuf, vbuf, eps6, out6);
  }

  // --- tail ---
  bn_relu_kernel<<<512, 256, 0, stream>>>(out6, gamma, beta, a6);
  fc1_wmma<<<dim3(1024 / 64, 128 / 32), 256, 0, stream>>>(a6, W1h, b1, a7);
  fc2_kernel<<<(1280 + 255) / 256, 256, 0, stream>>>(a7, W2, b2, (float*)d_out);
}
